// AttentionBlock_5506148073685
// MI455X (gfx1250) — matrix operand order repaired and checked by reading, not compiled, not run on hardware
//
#include <hip/hip_runtime.h>

// ---------------------------------------------------------------------------
// AttentionBlock for MI455X (gfx1250, wave32, WMMA).
// B=1, C=256, T=4096, 8 heads x 32 dims, GroupNorm(8).
// Flash-attention streaming (never materializes the 512MB TxT matrix);
// all math tiles via v_wmma_f32_16x16x32_bf16; everything L2-resident.
// ---------------------------------------------------------------------------

typedef __bf16 bf16_t;
typedef bf16_t v16bf __attribute__((ext_vector_type(16)));
typedef float  v8f   __attribute__((ext_vector_type(8)));
typedef unsigned short u16;

#define T_DIM 4096
#define C_DIM 256

__device__ __forceinline__ u16 f2bfbits(float f) {        // RNE f32 -> bf16 bits
  unsigned u = __builtin_bit_cast(unsigned, f);
  return (u16)((u + 0x7FFFu + ((u >> 16) & 1u)) >> 16);
}
__device__ __forceinline__ bf16_t bfb(u16 s) { return __builtin_bit_cast(bf16_t, s); }

// 16-bit A-matrix (16x32) k order used here (contiguous, as B) for lane-half hi (0:lanes0-15,1:lanes16-31)
__device__ __forceinline__ int a_kidx(int e, int hi) {
  return hi * 16 + e;
}

// ---------------- GroupNorm: x[c][t] -> xnT[t][c] (bf16) --------------------
__global__ __launch_bounds__(256) void gn_kernel(const float* __restrict__ x,
                                                 const float* __restrict__ gamma,
                                                 const float* __restrict__ beta,
                                                 u16* __restrict__ xnT) {
  __shared__ float rs[256], rq[256];
  const int g = blockIdx.x, tid = threadIdx.x;
  const float* xb = x + g * 32 * T_DIM;           // group = 32 channels x 4096
  float s = 0.f, q = 0.f;
  for (int i = tid; i < 32 * T_DIM; i += 256) { float v = xb[i]; s += v; q += v * v; }
  rs[tid] = s; rq[tid] = q; __syncthreads();
  for (int off = 128; off > 0; off >>= 1) {
    if (tid < off) { rs[tid] += rs[tid + off]; rq[tid] += rq[tid + off]; }
    __syncthreads();
  }
  const float inv_n = 1.0f / (32.0f * T_DIM);
  const float mean  = rs[0] * inv_n;
  const float var   = rq[0] * inv_n - mean * mean;
  const float istd  = rsqrtf(var + 1e-5f);
  for (int i = tid; i < 32 * T_DIM; i += 256) {
    int c = g * 32 + (i >> 12);
    int t = i & (T_DIM - 1);
    float v = (xb[i] - mean) * istd * gamma[c] + beta[c];
    xnT[t * C_DIM + c] = f2bfbits(v);
  }
}

// ---------------- fp32 -> bf16 weight convert -------------------------------
__global__ __launch_bounds__(256) void cvt_bf16_kernel(const float* __restrict__ s,
                                                       u16* __restrict__ d, int n) {
  int i = blockIdx.x * blockDim.x + threadIdx.x;
  if (i < n) d[i] = f2bfbits(s[i]);
}

// ---------------- QKV GEMM: [768x256]x[256x4096]+bias ----------------------
// Outputs (bf16): q[head*32+c][t] (scaled), kT[head][t][c] (scaled), v[head*32+c][t]
__global__ __launch_bounds__(128) void qkv_gemm_kernel(const u16* __restrict__ w,
                                                       const u16* __restrict__ xnT,
                                                       const float* __restrict__ bias,
                                                       u16* __restrict__ qb,
                                                       u16* __restrict__ kT,
                                                       u16* __restrict__ vb) {
  const int lane = threadIdx.x & 31, wv = threadIdx.x >> 5;
  const int hi = lane >> 4, N = lane & 15;
  const int n0 = (blockIdx.x * 4 + wv) * 16;     // t tile
  const int o0 = blockIdx.y * 16;                // output-channel tile
  v8f acc = {};
  for (int k0 = 0; k0 < C_DIM; k0 += 32) {
    v16bf a, b;
    const u16* ap = w + (o0 + N) * C_DIM + k0;   // row-major weights: 16 contiguous u16
    #pragma unroll
    for (int e = 0; e < 16; ++e) a[e] = bfb(ap[a_kidx(e, hi)]);
    const u16* bp = xnT + (n0 + N) * C_DIM + k0 + hi * 16;  // 16 contiguous u16
    #pragma unroll
    for (int e = 0; e < 16; ++e) b[e] = bfb(bp[e]);
    acc = __builtin_amdgcn_wmma_f32_16x16x32_bf16(false, a, false, b, (short)0, acc,
                                                  false, false);
  }
  const float qscale = 0.4204482076268573f;      // 32^(-1/4), applied to q and k
  #pragma unroll
  for (int v = 0; v < 8; ++v) {
    int o = o0 + v + hi * 8;
    int t = n0 + N;
    float val = acc[v] + bias[o];
    if (o < 256) {
      qb[o * T_DIM + t] = f2bfbits(val * qscale);
    } else if (o < 512) {
      int oo = o - 256, h = oo >> 5, c = oo & 31;
      kT[(h * T_DIM + t) * 32 + c] = f2bfbits(val * qscale);   // K transposed
    } else {
      vb[(o - 512) * T_DIM + t] = f2bfbits(val);
    }
  }
}

// ---------------- Flash attention per head, 16-query-row tile per wave ------
__global__ __launch_bounds__(128) void attn_kernel(const u16* __restrict__ qb,
                                                   const u16* __restrict__ kT,
                                                   const u16* __restrict__ vb,
                                                   u16* __restrict__ aT) {
  __shared__ u16 pst[4][16 * 32];                // per-wave P-tile staging (D->A layout)
  const int lane = threadIdx.x & 31, wv = threadIdx.x >> 5;
  const int hi = lane >> 4, N = lane & 15;
  const int head = blockIdx.y;
  const int qt0 = (blockIdx.x * 4 + wv) * 16;
  const u16* qh = qb + head * 32 * T_DIM;        // [c][t]
  const u16* kh = kT + head * T_DIM * 32;        // [t][c]
  const u16* vh = vb + head * 32 * T_DIM;        // [c][t]

  v16bf aq;                                      // Q^T tile: A[t=0..15][c=0..31]
  #pragma unroll
  for (int e = 0; e < 16; ++e) {
    int c = a_kidx(e, hi);
    aq[e] = bfb(qh[c * T_DIM + qt0 + N]);
  }

  float mrow[8], lsum[8];
  v8f acc0 = {}, acc1 = {};
  #pragma unroll
  for (int v = 0; v < 8; ++v) { mrow[v] = -3.0e38f; lsum[v] = 0.f; }
  u16* ps = &pst[wv][0];

  for (int s0 = 0; s0 < T_DIM; s0 += 32) {
    // ---- S = Q^T K : two 16x16 tiles, one WMMA each (K-depth 32 == head dim)
    v16bf bk0, bk1;
    const u16* kp0 = kh + (s0 + N) * 32 + hi * 16;      // 16 contiguous u16
    const u16* kp1 = kp0 + 16 * 32;
    __builtin_prefetch(kp0 + 32 * 32, 0, 0);            // next key tile
    #pragma unroll
    for (int e = 0; e < 16; ++e) { bk0[e] = bfb(kp0[e]); bk1[e] = bfb(kp1[e]); }
    v8f z = {};
    v8f S0 = __builtin_amdgcn_wmma_f32_16x16x32_bf16(false, aq, false, bk0, (short)0, z,
                                                     false, false);
    v8f S1 = __builtin_amdgcn_wmma_f32_16x16x32_bf16(false, aq, false, bk1, (short)0, z,
                                                     false, false);
    // ---- online softmax (row stats reduced across the 16-lane column group)
    #pragma unroll
    for (int v = 0; v < 8; ++v) {
      float x0 = S0[v], x1 = S1[v];
      float rmx = fmaxf(x0, x1);
      rmx = fmaxf(rmx, __shfl_xor(rmx, 1));
      rmx = fmaxf(rmx, __shfl_xor(rmx, 2));
      rmx = fmaxf(rmx, __shfl_xor(rmx, 4));
      rmx = fmaxf(rmx, __shfl_xor(rmx, 8));
      float mnew  = fmaxf(mrow[v], rmx);
      float alpha = __expf(mrow[v] - mnew);
      float p0 = __expf(x0 - mnew), p1 = __expf(x1 - mnew);
      float rsum = p0 + p1;
      rsum += __shfl_xor(rsum, 1);
      rsum += __shfl_xor(rsum, 2);
      rsum += __shfl_xor(rsum, 4);
      rsum += __shfl_xor(rsum, 8);
      lsum[v] = lsum[v] * alpha + rsum;
      mrow[v] = mnew;
      acc0[v] *= alpha;
      acc1[v] *= alpha;
      int r = v + hi * 8;                               // D layout row -> LDS row
      ps[r * 32 + N]      = f2bfbits(p0);
      ps[r * 32 + 16 + N] = f2bfbits(p1);
    }
    // ---- P tile back out of LDS in A-fragment layout (in-order DS, same wave)
    v16bf pa;
    #pragma unroll
    for (int e = 0; e < 16; ++e) pa[e] = bfb(ps[N * 32 + a_kidx(e, hi)]);
    // ---- O += P V^T : B-frags are contiguous along t in v[c][t]
    v16bf bv0, bv1;
    const u16* vp0 = vh + N * T_DIM + s0 + hi * 16;
    const u16* vp1 = vh + (16 + N) * T_DIM + s0 + hi * 16;
    __builtin_prefetch(vp0 + 32, 0, 0);
    #pragma unroll
    for (int e = 0; e < 16; ++e) { bv0[e] = bfb(vp0[e]); bv1[e] = bfb(vp1[e]); }
    acc0 = __builtin_amdgcn_wmma_f32_16x16x32_bf16(false, pa, false, bv0, (short)0, acc0,
                                                   false, false);
    acc1 = __builtin_amdgcn_wmma_f32_16x16x32_bf16(false, pa, false, bv1, (short)0, acc1,
                                                   false, false);
  }
  // ---- normalize and store transposed for the projection GEMM: aT[t][c]
  #pragma unroll
  for (int v = 0; v < 8; ++v) {
    float inv = 1.0f / lsum[v];
    int t = qt0 + v + hi * 8;
    aT[t * C_DIM + head * 32 + N]      = f2bfbits(acc0[v] * inv);
    aT[t * C_DIM + head * 32 + 16 + N] = f2bfbits(acc1[v] * inv);
  }
}

// ---------------- Projection GEMM + bias + residual (fp32 out) --------------
__global__ __launch_bounds__(128) void proj_gemm_kernel(const u16* __restrict__ w,
                                                        const u16* __restrict__ aT,
                                                        const float* __restrict__ bias,
                                                        const float* __restrict__ x,
                                                        float* __restrict__ out) {
  const int lane = threadIdx.x & 31, wv = threadIdx.x >> 5;
  const int hi = lane >> 4, N = lane & 15;
  const int n0 = (blockIdx.x * 4 + wv) * 16;
  const int o0 = blockIdx.y * 16;
  v8f acc = {};
  for (int k0 = 0; k0 < C_DIM; k0 += 32) {
    v16bf a, b;
    const u16* ap = w + (o0 + N) * C_DIM + k0;
    #pragma unroll
    for (int e = 0; e < 16; ++e) a[e] = bfb(ap[a_kidx(e, hi)]);
    const u16* bp = aT + (n0 + N) * C_DIM + k0 + hi * 16;
    #pragma unroll
    for (int e = 0; e < 16; ++e) b[e] = bfb(bp[e]);
    acc = __builtin_amdgcn_wmma_f32_16x16x32_bf16(false, a, false, b, (short)0, acc,
                                                  false, false);
  }
  #pragma unroll
  for (int v = 0; v < 8; ++v) {
    int o = o0 + v + hi * 8;
    int t = n0 + N;
    out[o * T_DIM + t] = x[o * T_DIM + t] + acc[v] + bias[o];
  }
}

// ---------------------------------------------------------------------------
extern "C" void kernel_launch(void* const* d_in, const int* in_sizes, int n_in,
                              void* d_out, int out_size, void* d_ws, size_t ws_size,
                              hipStream_t stream) {
  (void)in_sizes; (void)n_in; (void)out_size; (void)ws_size;
  const float* x      = (const float*)d_in[0];
  const float* gamma  = (const float*)d_in[1];
  const float* beta   = (const float*)d_in[2];
  const float* qkv_w  = (const float*)d_in[3];
  const float* qkv_b  = (const float*)d_in[4];
  const float* proj_w = (const float*)d_in[5];
  const float* proj_b = (const float*)d_in[6];
  float* out = (float*)d_out;

  // Workspace layout (~10.5 MB, all bf16 as u16 bits)
  char* ws = (char*)d_ws;
  u16* xnT = (u16*)(ws + 0);          // [4096][256]   2 MB
  u16* wq  = (u16*)(ws + 2097152);    // [768][256]    384 KB
  u16* wp  = (u16*)(ws + 2490368);    // [256][256]    128 KB
  u16* qb  = (u16*)(ws + 2621440);    // [256][4096]   2 MB (scaled q, [head*32+c][t])
  u16* kT  = (u16*)(ws + 4718592);    // [8][4096][32] 2 MB (scaled k, transposed)
  u16* vb  = (u16*)(ws + 6815744);    // [256][4096]   2 MB
  u16* aT  = (u16*)(ws + 8912896);    // [4096][256]   2 MB

  gn_kernel<<<8, 256, 0, stream>>>(x, gamma, beta, xnT);
  cvt_bf16_kernel<<<(768 * 256 + 255) / 256, 256, 0, stream>>>(qkv_w, wq, 768 * 256);
  cvt_bf16_kernel<<<(256 * 256 + 255) / 256, 256, 0, stream>>>(proj_w, wp, 256 * 256);
  qkv_gemm_kernel<<<dim3(64, 48), 128, 0, stream>>>(wq, xnT, qkv_b, qb, kT, vb);
  attn_kernel<<<dim3(64, 8), 128, 0, stream>>>(qb, kT, vb, aT);
  proj_gemm_kernel<<<dim3(64, 16), 128, 0, stream>>>(wp, aT, proj_b, x, out);
}
